// MTMFAAAC_35132832481494
// MI455X (gfx1250) — matrix operand order repaired and checked by reading, not compiled, not run on hardware
//
#include <hip/hip_runtime.h>

typedef __attribute__((ext_vector_type(16))) _Float16 v16h;
typedef __attribute__((ext_vector_type(8)))  float    v8f;

#define NAGENT 32
#define SDIM   128
#define ADIM   64
#define IDIM   192
#define HDIM   256
#define NOTH   31   // agents 1..31 participate in attention

// ---- workspace offsets (in halves) for pre-swizzled f16 weights ----
#define OFF_ENC 0L         // 32 * 192*256
#define OFF_S   1572864L   // 128*256
#define OFF_MF  1605632L   // 64*256
#define OFF_MF2 1622016L   // 64*256
#define OFF_K   1638400L   // 256*256
#define OFF_SEL 1703936L   // 256*256
#define OFF_V   1769472L   // 256*256
#define OFF_W1  1835008L   // 1024*256
#define OFF_W2  2097152L   // 256*64
// total = 2,113,536 halves = ~4.2 MB of d_ws

// ---- LDS strides (halves), padded for bank spread on A-fragment reads ----
#define CIN_S 1048   // critic-in 16 x 1024 (A source)
#define SAB_S 264    // sa / h1 tile 16 x 256 (A source)
#define XST_S 136    // state staging stride (K=128)
#define XMF_S 72     // meanfield staging stride (K=64)
#define XAG_S 200    // agent-input staging stride (K=192)

// Pre-swizzle f32 weights into the WMMA B-fragment layout, converted to f16.
// B tile (32K x 16N): lane = n within tile (lanes 0-15: K 0-7,16-23, lanes 16-31: K 8-15,24-31),
// 16 halves per lane contiguous. Tiles ordered [nt][kt], 512 halves each.
// src element (k,n) fetched as src[b*sbs + k*sk + (n/GN)*sg + (n%GN)*sn]
// (lets us flatten the (HEADS,H,AT) tensors to H x 256 without extra copies).
__global__ void swz_kernel(const float* __restrict__ src, _Float16* __restrict__ dst,
                           int K, int N, int GN, long sk, long sg, long sn,
                           long sbs, long dbs)
{
    long b = blockIdx.y;
    long e = (long)blockIdx.x * blockDim.x + threadIdx.x;
    long total = (long)K * N;
    if (e >= total) return;
    int tile   = (int)(e >> 9);
    int within = (int)(e & 511);
    int lane = within >> 4, h = within & 15;
    int KT = K >> 5;
    int nt = tile / KT, kt = tile - nt * KT;
    int n = nt * 16 + (lane & 15);
    int k = kt * 32 + ((lane >> 4) << 3) + (h & 7) + ((h >> 3) << 4);
    float v = src[b * sbs + (long)k * sk + (long)(n / GN) * sg + (long)(n % GN) * sn];
    dst[b * dbs + e] = (_Float16)v;
}

// Build a 16x32 f16 A fragment from an LDS row-major tile.
// ISA layout: lane m (0-15 dup in 16-31), VGPR p holds K pair; lanes>=16 get +8.
__device__ __forceinline__ v16h lds_a_frag(const _Float16* base, int strideH, int kbase, int lane)
{
    int m = lane & 15;
    int grp = (lane >> 4) * 8;
    const _Float16* row = base + m * strideH + kbase + grp;
    union { v16h v; unsigned u[8]; } r;
#pragma unroll
    for (int p = 0; p < 8; ++p) {
        int k = ((p < 4) ? 0 : 16) + (p & 3) * 2;   // even -> 4B-aligned pair load
        r.u[p] = *(const unsigned*)(row + k);
    }
    return r.v;
}

// One 16x16 output tile: C = act(A(16xK) * W(Kx16) + bias), A from LDS,
// W pre-swizzled in global (L2-resident), result stored f16 to LDS.
__device__ __forceinline__ void gemm16(const _Float16* __restrict__ aLds, int aStride, int K,
                                       const _Float16* __restrict__ wmat, int nt,
                                       const float* __restrict__ bias, bool act,
                                       _Float16* __restrict__ outLds, int outStride, int outColBase,
                                       int lane)
{
    int KT = K >> 5;
    v8f acc = {0.f,0.f,0.f,0.f,0.f,0.f,0.f,0.f};
    const _Float16* wt = wmat + (long)nt * KT * 512 + lane * 16;
    for (int kt = 0; kt < KT; ++kt) {
        v16h a = lds_a_frag(aLds, aStride, kt * 32, lane);
        v16h b = *(const v16h*)(wt + (long)kt * 512);
        acc = __builtin_amdgcn_wmma_f32_16x16x32_f16(false, a, false, b, (short)0, acc, false, false);
    }
    int c  = nt * 16 + (lane & 15);
    int rb = (lane >> 4) * 8;
    float bb = bias ? bias[c] : 0.f;
#pragma unroll
    for (int r = 0; r < 8; ++r) {
        float f = acc[r] + bb;
        if (act) f = f > 0.f ? f : 0.01f * f;   // leaky_relu(0.01)
        outLds[(rb + r) * outStride + outColBase + c] = (_Float16)f;
    }
}

__global__ void fused_kernel(
    const float* __restrict__ agent_inputs, const float* __restrict__ state,
    const float* __restrict__ mf, const float* __restrict__ mf2,
    const float* __restrict__ enc_b, const float* __restrict__ sb,
    const float* __restrict__ mfb, const float* __restrict__ mf2b,
    const float* __restrict__ bv, const float* __restrict__ b1, const float* __restrict__ b2,
    const _Float16* __restrict__ wp, float* __restrict__ out, int Bsz)
{
    extern __shared__ char smemraw[];
    _Float16* vals = (_Float16*)smemraw;          // 31 * 16*256
    _Float16* cin  = vals + NOTH * 16 * HDIM;     // 16 * CIN_S  (cols: senc|mf|mf2|other)
    _Float16* selb = cin  + 16 * CIN_S;           // 16 * 256
    _Float16* sab  = selb + 16 * HDIM;            // 16 * SAB_S
    _Float16* xab  = sab  + 16 * SAB_S;           // 16 * 256 staging / keys
    float*    lg   = (float*)(xab + 16 * HDIM);   // 4 * 16 * 32 logits/weights

    const _Float16* wp_enc = wp + OFF_ENC;
    const _Float16* wp_s   = wp + OFF_S;
    const _Float16* wp_mf  = wp + OFF_MF;
    const _Float16* wp_mf2 = wp + OFF_MF2;
    const _Float16* wp_k   = wp + OFF_K;
    const _Float16* wp_sel = wp + OFF_SEL;
    const _Float16* wp_v   = wp + OFF_V;
    const _Float16* wp_w1  = wp + OFF_W1;
    const _Float16* wp_w2  = wp + OFF_W2;

    const int t = threadIdx.x;
    const int lane = t & 31;
    const int wave = t >> 5;
    const long b0 = (long)blockIdx.x * 16;

    // ---- state encoder: s_enc -> cin[:,0:256] ----
    for (int i = t; i < 16 * SDIM; i += 256) {
        int r = i >> 7, c = i & 127;
        xab[r * XST_S + c] = (_Float16)state[b0 * SDIM + i];
    }
    __syncthreads();
    for (int j = 0; j < 2; ++j)
        gemm16(xab, XST_S, SDIM, wp_s, wave * 2 + j, sb, true, cin, CIN_S, 0, lane);
    __syncthreads();
    // ---- sel = s_enc @ Wsel (no bias) ----
    for (int j = 0; j < 2; ++j)
        gemm16(cin, CIN_S, HDIM, wp_sel, wave * 2 + j, nullptr, false, selb, HDIM, 0, lane);
    __syncthreads();
    // ---- meanfield encoders -> cin[:,256:512], cin[:,512:768] ----
    for (int i = t; i < 16 * ADIM; i += 256) {
        int r = i >> 6, c = i & 63;
        xab[r * XMF_S + c] = (_Float16)mf[b0 * ADIM + i];
    }
    __syncthreads();
    for (int j = 0; j < 2; ++j)
        gemm16(xab, XMF_S, ADIM, wp_mf, wave * 2 + j, mfb, true, cin, CIN_S, HDIM, lane);
    __syncthreads();
    for (int i = t; i < 16 * ADIM; i += 256) {
        int r = i >> 6, c = i & 63;
        xab[r * XMF_S + c] = (_Float16)mf2[b0 * ADIM + i];
    }
    __syncthreads();
    for (int j = 0; j < 2; ++j)
        gemm16(xab, XMF_S, ADIM, wp_mf2, wave * 2 + j, mf2b, true, cin, CIN_S, 2 * HDIM, lane);
    __syncthreads();

    // ---- per-agent: encoder -> keys/vals -> logits (agent 0 is dead) ----
    for (int n = 1; n < NAGENT; ++n) {
        const float* Xn = agent_inputs + (long)n * Bsz * IDIM + b0 * IDIM;
        for (int i = t; i < 16 * IDIM; i += 256) {
            int r = i / IDIM, c = i - r * IDIM;
            xab[r * XAG_S + c] = (_Float16)Xn[i];
        }
        __syncthreads();
        for (int j = 0; j < 2; ++j)
            gemm16(xab, XAG_S, IDIM, wp_enc + (long)n * IDIM * HDIM, wave * 2 + j,
                   enc_b + n * HDIM, true, sab, SAB_S, 0, lane);
        __syncthreads();
        for (int j = 0; j < 2; ++j) {
            gemm16(sab, SAB_S, HDIM, wp_k, wave * 2 + j, nullptr, false, xab, HDIM, 0, lane);
            gemm16(sab, SAB_S, HDIM, wp_v, wave * 2 + j, bv, true,
                   vals + (long)(n - 1) * 16 * HDIM, HDIM, 0, lane);
        }
        __syncthreads();
        if (t < 64) {   // logits[e,b,n-1] = sel . keys / sqrt(64)
            int e = t >> 4, r = t & 15;
            const _Float16* sp = selb + r * HDIM + e * 64;
            const _Float16* kp = xab  + r * HDIM + e * 64;
            float s = 0.f;
            for (int d = 0; d < 64; ++d) s += (float)sp[d] * (float)kp[d];
            lg[(e * 16 + r) * 32 + (n - 1)] = s * 0.125f;
        }
        __syncthreads();
    }

    // ---- softmax over the 31 agents ----
    if (t < 64) {
        int e = t >> 4, r = t & 15;
        float* row = lg + (e * 16 + r) * 32;
        float m = row[0];
        for (int i = 1; i < NOTH; ++i) m = fmaxf(m, row[i]);
        float s = 0.f;
        for (int i = 0; i < NOTH; ++i) { float x = __expf(row[i] - m); row[i] = x; s += x; }
        float inv = 1.f / s;
        for (int i = 0; i < NOTH; ++i) row[i] *= inv;
    }
    __syncthreads();

    // ---- other_flat = sum_n w * vals -> cin[:,768:1024] ----
    for (int i = t; i < 16 * HDIM; i += 256) {
        int r = i >> 8, c = i & 255;
        const float* w = lg + (((c >> 6) * 16 + r) * 32);
        float s = 0.f;
        for (int n2 = 0; n2 < NOTH; ++n2)
            s += w[n2] * (float)vals[(long)n2 * 16 * HDIM + r * HDIM + c];
        cin[r * CIN_S + 3 * HDIM + c] = (_Float16)s;
    }
    __syncthreads();

    // ---- critic: h1 = lrelu(cin @ W1 + b1) ----
    for (int j = 0; j < 2; ++j)
        gemm16(cin, CIN_S, 1024, wp_w1, wave * 2 + j, b1, true, sab, SAB_S, 0, lane);
    __syncthreads();
    // ---- out = h1 @ W2 + b2 (64 cols -> waves 0..3), f32 to global ----
    if (wave < 4) {
        v8f acc = {0.f,0.f,0.f,0.f,0.f,0.f,0.f,0.f};
        const _Float16* wt = wp_w2 + (long)wave * 8 * 512 + lane * 16;
        for (int kt = 0; kt < 8; ++kt) {
            v16h a = lds_a_frag(sab, SAB_S, kt * 32, lane);
            v16h b = *(const v16h*)(wt + (long)kt * 512);
            acc = __builtin_amdgcn_wmma_f32_16x16x32_f16(false, a, false, b, (short)0, acc, false, false);
        }
        int c  = wave * 16 + (lane & 15);
        int rb = (lane >> 4) * 8;
        float bb = b2[c];
#pragma unroll
        for (int r = 0; r < 8; ++r)
            out[(b0 + rb + r) * 64 + c] = acc[r] + bb;
    }
}

extern "C" void kernel_launch(void* const* d_in, const int* in_sizes, int n_in,
                              void* d_out, int out_size, void* d_ws, size_t ws_size,
                              hipStream_t stream)
{
    (void)n_in; (void)out_size; (void)ws_size;
    const float* agent_inputs = (const float*)d_in[0];
    const float* state  = (const float*)d_in[1];
    const float* mf     = (const float*)d_in[2];
    const float* mf2    = (const float*)d_in[3];
    const float* enc_W  = (const float*)d_in[4];
    const float* enc_b  = (const float*)d_in[5];
    const float* sW     = (const float*)d_in[6];
    const float* sb     = (const float*)d_in[7];
    const float* mfW    = (const float*)d_in[8];
    const float* mfb    = (const float*)d_in[9];
    const float* mf2W   = (const float*)d_in[10];
    const float* mf2b   = (const float*)d_in[11];
    const float* Wk     = (const float*)d_in[12];
    const float* Wsel   = (const float*)d_in[13];
    const float* Wv     = (const float*)d_in[14];
    const float* bv     = (const float*)d_in[15];
    const float* W1     = (const float*)d_in[16];
    const float* b1     = (const float*)d_in[17];
    const float* W2     = (const float*)d_in[18];
    const float* b2     = (const float*)d_in[19];
    _Float16* wp  = (_Float16*)d_ws;
    float*    out = (float*)d_out;

    const int Bsz = in_sizes[1] / SDIM;   // 16384

    auto swz = [&](const float* src, long off, int K, int N, int GN,
                   long sk, long sg, long sn, int batch, long sbs, long dbs) {
        dim3 g((unsigned)((K * N + 255) / 256), (unsigned)batch);
        swz_kernel<<<g, 256, 0, stream>>>(src, wp + off, K, N, GN, sk, sg, sn, sbs, dbs);
    };
    // plain K x N matrices: sk=N, sn=1, GN=N (sg unused)
    swz(enc_W, OFF_ENC, 192, 256, 256, 256, 0, 1, 32, 192L * 256, 192L * 256);
    swz(sW,    OFF_S,   128, 256, 256, 256, 0, 1, 1, 0, 0);
    swz(mfW,   OFF_MF,   64, 256, 256, 256, 0, 1, 1, 0, 0);
    swz(mf2W,  OFF_MF2,  64, 256, 256, 256, 0, 1, 1, 0, 0);
    // (HEADS,H,AT) tensors flattened to 256 x 256 with col = e*64+d
    swz(Wk,    OFF_K,   256, 256, 64, 64, 256L * 64, 1, 1, 0, 0);
    swz(Wsel,  OFF_SEL, 256, 256, 64, 64, 256L * 64, 1, 1, 0, 0);
    swz(Wv,    OFF_V,   256, 256, 64, 64, 256L * 64, 1, 1, 0, 0);
    swz(W1,    OFF_W1, 1024, 256, 256, 256, 0, 1, 1, 0, 0);
    swz(W2,    OFF_W2,  256,  64,  64,  64, 0, 1, 1, 0, 0);

    // dynamic LDS: vals + cin + selb + sab + xab (halves) + logits (floats)
    size_t lds_bytes =
        (size_t)(NOTH * 16 * HDIM + 16 * CIN_S + 16 * HDIM + 16 * SAB_S + 16 * HDIM) * 2
        + (size_t)4 * 16 * 32 * 4;   // = 320,512 B (< 320 KB WGP LDS)

    fused_kernel<<<dim3(Bsz / 16), 256, lds_bytes, stream>>>(
        agent_inputs, state, mf, mf2, enc_b, sb, mfb, mf2b, bv, b1, b2, wp, out, Bsz);
}
